// QwenSelfAttention_32555852104039
// MI455X (gfx1250) — matrix operand order repaired and checked by reading, not compiled, not run on hardware
//
#include <hip/hip_runtime.h>
#include <hip/hip_bf16.h>

// ---------------------------------------------------------------------------
// Problem constants (match reference)
// ---------------------------------------------------------------------------
constexpr int BB  = 2;
constexpr int TT  = 2048;
constexpr int CC  = 2048;
constexpr int HH  = 16;
constexpr int HKK = 8;
constexpr int DD  = 128;    // head dim
constexpr int NQKV = HH * DD + 2 * HKK * DD;   // 4096 fused QKV columns
constexpr float NEGF = -3.402823466e38f;
constexpr float SM_SCALE = 0.08838834764831845f; // 1/sqrt(128)

// ---------------------------------------------------------------------------
// Vector / fragment types
// ---------------------------------------------------------------------------
typedef __attribute__((ext_vector_type(8)))  float        v8f;
typedef __attribute__((ext_vector_type(16))) __bf16       v16bf;
typedef __attribute__((ext_vector_type(4)))  unsigned int u32x4;
typedef __attribute__((ext_vector_type(4)))  float        f32x4;
typedef __attribute__((ext_vector_type(4)))  unsigned int v4u;
typedef __attribute__((ext_vector_type(4)))  int          v4i;
typedef __attribute__((ext_vector_type(8)))  int          v8i;

union FragBF {              // 32 bytes: one WMMA A/B operand (16 bf16)
    v16bf  v;
    __bf16 e[16];
    u32x4  q[2];
};

__device__ __forceinline__ v8f wmma_bf16(const FragBF& a, const FragBF& b, v8f c) {
    return __builtin_amdgcn_wmma_f32_16x16x32_bf16(
        /*neg_a=*/false, a.v, /*neg_b=*/false, b.v,
        /*c_mod=*/(short)0, c, /*reuse_a=*/false, /*reuse_b=*/false);
}

// ---------------------------------------------------------------------------
// Tensor Data Mover: 2D tile load Global -> LDS (CDNA5 ISA ch.8 D# layout).
// data_size = 2B; LDS padding: +4 DWORDs every 16 DWORDs => 64B tile row is
// stored with 80B stride (16B aligned, conflict-free: 20*r mod 64 distinct).
// ---------------------------------------------------------------------------
__device__ __forceinline__ unsigned lds_addr_of(const void* p) {
    return (unsigned)(size_t)(__attribute__((address_space(3))) const void*)p;
}

__device__ __forceinline__ void tdm_load_tile_2d(unsigned lds_off, const void* gptr,
                                                 unsigned tensor_w, unsigned tensor_h,
                                                 unsigned tile_w, unsigned tile_h,
                                                 unsigned row_stride_elems) {
    unsigned long long ga = (unsigned long long)(size_t)gptr;
    v4u g0;
    g0[0] = 1u;                                        // count=1 (valid user D#)
    g0[1] = lds_off;                                   // lds_addr (bytes)
    g0[2] = (unsigned)(ga & 0xffffffffu);              // global_addr[31:0]
    g0[3] = (unsigned)((ga >> 32) & 0x01ffffffu)       // global_addr[56:32]
          | (2u << 30);                                // type = 2 ("image")
    v8i g1;
    g1[0] = (int)((1u << 16)                           // data_size = 1 -> 2 bytes
                | (1u << 20)                           // pad_enable
                | (3u << 22)                           // pad_interval: 16 DWORDs
                | (3u << 25));                         // pad_amount:   4 DWORDs
    g1[1] = (int)((tensor_w & 0xffffu) << 16);         // tensor_dim0[15:0]
    g1[2] = (int)(((tensor_w >> 16) & 0xffffu)         // tensor_dim0[31:16]
                | ((tensor_h & 0xffffu) << 16));       // tensor_dim1[15:0]
    g1[3] = (int)(((tensor_h >> 16) & 0xffffu)         // tensor_dim1[31:16]
                | ((tile_w & 0xffffu) << 16));         // tile_dim0
    g1[4] = (int)(tile_h & 0xffffu);                   // tile_dim1 (tile_dim2=0)
    g1[5] = (int)row_stride_elems;                     // tensor_dim0_stride[31:0]
    g1[6] = 0;                                         // stride hi / dim1_stride lo
    g1[7] = 0;
    v4i z4 = {0, 0, 0, 0};                             // <=2D: groups 2/3 unused
    v8i z8 = {0, 0, 0, 0, 0, 0, 0, 0};
    // amdgpu-toolchain (clang-23) 6-arg form: (g0, g1, g2, g3, extra, cpol)
    __builtin_amdgcn_tensor_load_to_lds(g0, g1, z4, z4, z8, 0);
}

// ---------------------------------------------------------------------------
// Stage 1: precision conversion / weight packing (weights stored TRANSPOSED
// [N][K] so GEMM B-tiles are plain 2D row-tiles for the TDM)
// ---------------------------------------------------------------------------
__global__ __launch_bounds__(256)
void cvt_f32_to_bf16(const float* __restrict__ src, __bf16* __restrict__ dst, int n) {
    int i = (blockIdx.x * 256 + threadIdx.x) * 4;
    if (i + 3 < n) {
        f32x4 v = *reinterpret_cast<const f32x4*>(src + i);
        dst[i + 0] = (__bf16)v[0];
        dst[i + 1] = (__bf16)v[1];
        dst[i + 2] = (__bf16)v[2];
        dst[i + 3] = (__bf16)v[3];
    } else {
        for (int k = i; k < n; ++k) dst[k] = (__bf16)src[k];
    }
}

__global__ __launch_bounds__(256)
void pack_wqkv_t_kernel(const float* __restrict__ Wq, const float* __restrict__ Wk,
                        const float* __restrict__ Wv, __bf16* __restrict__ dst) {
    int idx = blockIdx.x * 256 + threadIdx.x;      // over NQKV * CC
    int k = idx & (CC - 1);
    int n = idx >> 11;                              // CC == 2048
    float v;
    if (n < HH * DD)                 v = Wq[(size_t)k * (HH * DD) + n];
    else if (n < HH * DD + HKK * DD) v = Wk[(size_t)k * (HKK * DD) + (n - HH * DD)];
    else                             v = Wv[(size_t)k * (HKK * DD) + (n - HH * DD - HKK * DD)];
    dst[idx] = (__bf16)v;
}

__global__ __launch_bounds__(256)
void cvt_transpose_bf16(const float* __restrict__ src, __bf16* __restrict__ dst) {
    int idx = blockIdx.x * 256 + threadIdx.x;      // over CC * CC
    int k = idx & (CC - 1);
    int n = idx >> 11;
    dst[idx] = (__bf16)src[(size_t)k * CC + n];
}

// ---------------------------------------------------------------------------
// WMMA GEMM:  C[M,N] = A[M,K] * Bt[N,K]^T   (A row-major, Bt = B transposed)
// Block: 256 threads = 8 waves; block tile 128x128; wave tile 32x64.
// Tile staging is done by the Tensor Data Mover (wave 0 issues two 128x32
// tile DMAs, drains TENSORcnt, block barrier publishes). LDS tiles land
// row-major with 80B padded stride; fragment reads are 2x ds_load_b128.
// ---------------------------------------------------------------------------
template <typename OutT>
__global__ __launch_bounds__(256)
void gemm_bf16_kernel(const __bf16* __restrict__ A, const __bf16* __restrict__ Bt,
                      OutT* __restrict__ Cm, int M, int N, int K) {
    __shared__ __align__(16) __bf16 As[128][40];   // 128 x 32 (+4 DW TDM pad)
    __shared__ __align__(16) __bf16 Bs[128][40];   // 128 x 32 (+4 DW TDM pad)

    const int tid = threadIdx.x;
    const int lane = tid & 31, w = tid >> 5;
    const int lane16 = lane & 15, hi = lane >> 4;
    const int wm = w >> 1, wn = w & 1;             // 4x2 wave grid
    const int m0 = blockIdx.y * 128, n0 = blockIdx.x * 128;

    v8f acc[2][4];
    #pragma unroll
    for (int mi = 0; mi < 2; ++mi)
        #pragma unroll
        for (int ni = 0; ni < 4; ++ni) acc[mi][ni] = {};

    const unsigned ldsA = lds_addr_of(&As[0][0]);
    const unsigned ldsB = lds_addr_of(&Bs[0][0]);

    for (int k0 = 0; k0 < K; k0 += 32) {
        asm volatile("s_wait_dscnt 0" ::: "memory");   // prior-tile reads drained
        __syncthreads();
        if (w == 0) {   // one DMA per tensor instruction (TDM ignores EXEC)
            tdm_load_tile_2d(ldsA, A  + (size_t)m0 * K + k0,
                             (unsigned)K, (unsigned)M, 32u, 128u, (unsigned)K);
            tdm_load_tile_2d(ldsB, Bt + (size_t)n0 * K + k0,
                             (unsigned)K, (unsigned)N, 32u, 128u, (unsigned)K);
            __builtin_amdgcn_s_wait_tensorcnt(0);
        }
        __syncthreads();

        // Fragments from row-major tile: two 16B runs at {hi*8, hi*8+16}
        FragBF af[2];
        #pragma unroll
        for (int mi = 0; mi < 2; ++mi) {
            const __bf16* p = &As[wm * 32 + mi * 16 + lane16][hi * 8];
            af[mi].q[0] = *reinterpret_cast<const u32x4*>(p);
            af[mi].q[1] = *reinterpret_cast<const u32x4*>(p + 16);
        }
        #pragma unroll
        for (int ni = 0; ni < 4; ++ni) {
            FragBF bf;
            const __bf16* p = &Bs[wn * 64 + ni * 16 + lane16][hi * 8];
            bf.q[0] = *reinterpret_cast<const u32x4*>(p);
            bf.q[1] = *reinterpret_cast<const u32x4*>(p + 16);
            #pragma unroll
            for (int mi = 0; mi < 2; ++mi)
                acc[mi][ni] = wmma_bf16(af[mi], bf, acc[mi][ni]);
        }
    }

    // Epilogue: C layout -> row = (r + 8*hi), col = lane16 within each 16x16
    #pragma unroll
    for (int mi = 0; mi < 2; ++mi)
        #pragma unroll
        for (int ni = 0; ni < 4; ++ni) {
            int col = n0 + wn * 64 + ni * 16 + lane16;
            #pragma unroll
            for (int r = 0; r < 8; ++r) {
                int row = m0 + wm * 32 + mi * 16 + hi * 8 + r;
                Cm[(size_t)row * N + col] = (OutT)(acc[mi][ni][r]);
            }
        }
}

// ---------------------------------------------------------------------------
// RoPE + head split: qkv[B*T,4096] -> Q[B,H,T,D], K[B,HK,T,D], Vt[B,HK,D,T]
// ---------------------------------------------------------------------------
__global__ __launch_bounds__(256)
void rope_kernel(const __bf16* __restrict__ qkv, const int* __restrict__ pos,
                 __bf16* __restrict__ Q, __bf16* __restrict__ K, __bf16* __restrict__ Vt) {
    int idx = blockIdx.x * 256 + threadIdx.x;  // 2^22 threads total
    int j = idx & 63;
    int h = (idx >> 6) & 15;
    int t = (idx >> 10) & 2047;
    int b = idx >> 21;
    if (b >= BB) return;

    float p = (float)pos[t];
    float inv_theta = __powf(10000.0f, -(float)(2 * j) / (float)DD);
    float f = p * inv_theta;
    float c = __cosf(f), s = __sinf(f);

    const __bf16* row = qkv + (size_t)(b * TT + t) * NQKV;
    {   // Q head h
        float x1 = (float)row[h * DD + j];
        float x2 = (float)row[h * DD + 64 + j];
        __bf16* qd = Q + ((size_t)(b * HH + h) * TT + t) * DD;
        qd[j]      = (__bf16)(x1 * c - x2 * s);
        qd[64 + j] = (__bf16)(x2 * c + x1 * s);
    }
    if (h < HKK) {   // K head h (rotated) + V head h (transposed copy)
        float x1 = (float)row[HH * DD + h * DD + j];
        float x2 = (float)row[HH * DD + h * DD + 64 + j];
        __bf16* kd = K + ((size_t)(b * HKK + h) * TT + t) * DD;
        kd[j]      = (__bf16)(x1 * c - x2 * s);
        kd[64 + j] = (__bf16)(x2 * c + x1 * s);
        __bf16* vt = Vt + (size_t)(b * HKK + h) * DD * TT;
        vt[(size_t)j * TT + t]        = row[HH * DD + HKK * DD + h * DD + j];
        vt[(size_t)(64 + j) * TT + t] = row[HH * DD + HKK * DD + h * DD + 64 + j];
    }
}

// ---------------------------------------------------------------------------
// Flash attention (causal, GQA G=2). 128 threads = 4 waves per block;
// each wave owns one 16-query tile. Online softmax over 32-key chunks.
// K and Vt fragments load straight from global (contiguous, 2x b128 each);
// only the C->A layout conversion of P goes through LDS (fragment-ordered).
// ---------------------------------------------------------------------------
__global__ __launch_bounds__(128)
void attn_kernel(const __bf16* __restrict__ Q, const __bf16* __restrict__ K,
                 const __bf16* __restrict__ Vt, const unsigned char* __restrict__ mask,
                 __bf16* __restrict__ O) {
    __shared__ __align__(16) __bf16 Ps[4][16][40];   // per-wave P tile, fragment order

    const int tid = threadIdx.x;
    const int lane = tid & 31, w = tid >> 5;
    const int lane16 = lane & 15, hi = lane >> 4;

    int gw = blockIdx.x * 4 + w;
    const int qtiles = TT / 16;
    int qt  = gw % qtiles;
    int bh  = gw / qtiles;
    int h   = bh % HH;
    int b   = bh / HH;
    int kvh = h >> 1;                  // G = H/HK = 2
    int q0  = qt * 16;

    const __bf16* qrow = Q + ((size_t)(b * HH + h) * TT + (q0 + lane16)) * DD;
    FragBF qa[4];
    #pragma unroll
    for (int c = 0; c < 4; ++c) {
        const u32x4* p = reinterpret_cast<const u32x4*>(qrow + c * 32 + hi * 16);
        qa[c].q[0] = p[0];
        qa[c].q[1] = p[1];     // +8 elements = +16B
    }

    const __bf16* kbase  = K  + (size_t)(b * HKK + kvh) * TT * DD;
    const __bf16* vtbase = Vt + (size_t)(b * HKK + kvh) * DD * TT;
    const unsigned char* mrow = mask + (size_t)b * TT;

    // Fragment-order positions for P writes (A-fragment permutation).
    const int pk0 = lane16;                                  // k = lane16
    const int pk1 = 16 + lane16;                             // k = 16 + lane16

    float m_r[8], l_r[8];
    v8f o[8];
    #pragma unroll
    for (int r = 0; r < 8; ++r) { m_r[r] = NEGF; l_r[r] = 0.0f; }
    #pragma unroll
    for (int ni = 0; ni < 8; ++ni) o[ni] = {};

    const int nch = (q0 + 47) / 32;          // key chunks covering keys <= q0+15
    for (int kc = 0; kc < nch; ++kc) {
        const int key0 = kc * 32;

        // Scores: S(16x32) = Q(16x128) x K^T; K fragments contiguous in d.
        v8f s0 = {}, s1 = {};
        #pragma unroll
        for (int c = 0; c < 4; ++c) {
            FragBF kb0, kb1;
            const u32x4* k0p = reinterpret_cast<const u32x4*>(
                kbase + (size_t)(key0 + lane16) * DD + c * 32 + hi * 16);
            const u32x4* k1p = reinterpret_cast<const u32x4*>(
                kbase + (size_t)(key0 + 16 + lane16) * DD + c * 32 + hi * 16);
            kb0.q[0] = k0p[0]; kb0.q[1] = k0p[1];
            kb1.q[0] = k1p[0]; kb1.q[1] = k1p[1];
            s0 = wmma_bf16(qa[c], kb0, s0);
            s1 = wmma_bf16(qa[c], kb1, s1);
        }

        // Online softmax. C-layout: element r -> row q0 + 8*hi + r, col lane16.
        #pragma unroll
        for (int r = 0; r < 8; ++r) {
            int row = q0 + 8 * hi + r;
            int kk0 = key0 + lane16;
            int kk1 = key0 + 16 + lane16;
            float v0 = s0[r] * SM_SCALE;
            float v1 = s1[r] * SM_SCALE;
            if (kk0 > row || !mrow[kk0]) v0 = NEGF;
            if (kk1 > row || !mrow[kk1]) v1 = NEGF;
            float rm = fmaxf(v0, v1);
            rm = fmaxf(rm, __shfl_xor(rm, 1, 32));
            rm = fmaxf(rm, __shfl_xor(rm, 2, 32));
            rm = fmaxf(rm, __shfl_xor(rm, 4, 32));
            rm = fmaxf(rm, __shfl_xor(rm, 8, 32));
            float mn   = fmaxf(m_r[r], rm);
            float corr = __expf(m_r[r] - mn);
            float p0   = __expf(v0 - mn);
            float p1   = __expf(v1 - mn);
            float rs   = p0 + p1;
            rs += __shfl_xor(rs, 1, 32);
            rs += __shfl_xor(rs, 2, 32);
            rs += __shfl_xor(rs, 4, 32);
            rs += __shfl_xor(rs, 8, 32);
            l_r[r] = l_r[r] * corr + rs;
            m_r[r] = mn;
            #pragma unroll
            for (int ni = 0; ni < 8; ++ni) o[ni][r] *= corr;
            Ps[w][8 * hi + r][pk0] = (__bf16)p0;
            Ps[w][8 * hi + r][pk1] = (__bf16)p1;
        }

        // Wave-local LDS RAW: same-wave DS ops are in-order; drain them.
        asm volatile("s_wait_dscnt 0" ::: "memory");

        // P as A-fragment: fragment-ordered rows -> 2x ds_load_b128.
        FragBF pf;
        {
            const u32x4* p = reinterpret_cast<const u32x4*>(&Ps[w][lane16][hi * 16]);
            pf.q[0] = p[0];
            pf.q[1] = p[1];
        }
        // O(16x128) += P(16x32) x V(32x128); Vt fragments contiguous in T.
        #pragma unroll
        for (int ni = 0; ni < 8; ++ni) {
            FragBF vb;
            const u32x4* p = reinterpret_cast<const u32x4*>(
                vtbase + (size_t)(ni * 16 + lane16) * TT + key0 + hi * 16);
            vb.q[0] = p[0];
            vb.q[1] = p[1];
            o[ni] = wmma_bf16(pf, vb, o[ni]);
        }
    }

    // Normalize and store attn output as [B, T, H*D] bf16 (GEMM-ready).
    #pragma unroll
    for (int r = 0; r < 8; ++r) {
        float inv = 1.0f / l_r[r];
        int t = q0 + 8 * hi + r;
        __bf16* orow = O + (size_t)(b * TT + t) * (HH * DD) + h * DD;
        #pragma unroll
        for (int ni = 0; ni < 8; ++ni)
            orow[ni * 16 + lane16] = (__bf16)(o[ni][r] * inv);
    }
}

// ---------------------------------------------------------------------------
// Launcher. Workspace layout (bytes), liveness-based aliasing (72 MB total):
//   [ 0,16M)  xbf            -> later Q  [B,H,T,D]
//   [16M,32M) wqkv^T         -> later K [B,HK,T,D] | Vt [B,HK,D,T]
//   [32M,40M) Wo^T bf16 (persists)
//   [40M,72M) qkv bf16 [B*T,4096] -> later attn-out [B*T,2048]
// ---------------------------------------------------------------------------
extern "C" void kernel_launch(void* const* d_in, const int* in_sizes, int n_in,
                              void* d_out, int out_size, void* d_ws, size_t ws_size,
                              hipStream_t stream) {
    const float*         x    = (const float*)d_in[0];
    const unsigned char* mask = (const unsigned char*)d_in[1];
    const int*           pos  = (const int*)d_in[2];
    const float*         Wq   = (const float*)d_in[3];
    const float*         Wk   = (const float*)d_in[4];
    const float*         Wv   = (const float*)d_in[5];
    const float*         Wo   = (const float*)d_in[6];

    char* ws = (char*)d_ws;
    const size_t MB16 = 16ull << 20;
    __bf16* xbf  = (__bf16*)(ws + 0);
    __bf16* wqkv = (__bf16*)(ws + MB16);                  // transposed [4096][2048]
    __bf16* wob  = (__bf16*)(ws + 2 * MB16);              // transposed [2048][2048]
    __bf16* qkv  = (__bf16*)(ws + 2 * MB16 + (8ull << 20));
    // aliases (regions dead by the time these are written)
    __bf16* Qb   = xbf;
    __bf16* Kb   = (__bf16*)(ws + MB16);
    __bf16* Vtb  = (__bf16*)(ws + MB16 + (8ull << 20));
    __bf16* attn = qkv;

    const int M = BB * TT;   // 4096

    // 1) precision staging (weights transposed -> TDM-friendly row tiles)
    cvt_f32_to_bf16<<<(M * CC) / 1024, 256, 0, stream>>>(x, xbf, M * CC);
    pack_wqkv_t_kernel<<<(NQKV * CC) / 256, 256, 0, stream>>>(Wq, Wk, Wv, wqkv);
    cvt_transpose_bf16<<<(CC * CC) / 256, 256, 0, stream>>>(Wo, wob);

    // 2) fused QKV projection (bf16 WMMA, f32 accum, TDM tile staging)
    dim3 gq(NQKV / 128, M / 128);
    gemm_bf16_kernel<__bf16><<<gq, 256, 0, stream>>>(xbf, wqkv, qkv, M, NQKV, CC);

    // 3) RoPE + head transpose (V stored transposed)
    rope_kernel<<<(BB * TT * HH * 64) / 256, 256, 0, stream>>>(qkv, pos, Qb, Kb, Vtb);

    // 4) causal GQA flash attention
    attn_kernel<<<(BB * HH * (TT / 16)) / 4, 128, 0, stream>>>(Qb, Kb, Vtb, mask, attn);

    // 5) output projection -> f32 d_out
    dim3 go(CC / 128, M / 128);
    gemm_bf16_kernel<float><<<go, 256, 0, stream>>>(attn, wob, (float*)d_out, M, CC, CC);
}
